// Conv2dAttention_77360950935554
// MI455X (gfx1250) — matrix operand order repaired and checked by reading, not compiled, not run on hardware
//
#include <hip/hip_runtime.h>

// ---------------------------------------------------------------------------
// Conv2d self-attention for MI455X (gfx1250), all GEMM stages via
// v_wmma_f32_16x16x32_bf16 (wave32 WMMA, f32 accumulate).
// ---------------------------------------------------------------------------

typedef __attribute__((ext_vector_type(16))) __bf16 v16bf;
typedef __attribute__((ext_vector_type(8)))  __bf16 v8bf;
typedef __attribute__((ext_vector_type(8)))  float  v8f;

#define WMMA_BF16(A, B, C)                                                    \
  __builtin_amdgcn_wmma_f32_16x16x32_bf16(false, (A), false, (B), (short)0,   \
                                          (C), false, false)

static constexpr int kB  = 16;
static constexpr int kC  = 256;
static constexpr int kL  = 1024;   // H*W
static constexpr int kNH = 8;
static constexpr int kDH = 32;

// Build a v16bf A-fragment from two contiguous 8-element halves.
__device__ __forceinline__ v16bf make_afrag(const __bf16* p0, const __bf16* p1) {
  v8bf lo = *(const v8bf*)p0;
  v8bf hi = *(const v8bf*)p1;
  v16bf a;
#pragma unroll
  for (int e = 0; e < 8; ++e) { a[e] = lo[e]; a[8 + e] = hi[e]; }
  return a;
}

// ---------------------------------------------------------------------------
// Kernel 1: x [b][c][l] f32  ->  xb_t [b][l][c] bf16 (transpose + convert)
// ---------------------------------------------------------------------------
__global__ void cvt_x_kernel(const float* __restrict__ x,
                             __bf16* __restrict__ xb_t) {
  size_t i = (size_t)blockIdx.x * blockDim.x + threadIdx.x; // over B*L*C
  int c = (int)(i % kC);
  size_t t = i / kC;
  int l = (int)(t % kL);
  int b = (int)(t / kL);
  xb_t[i] = (__bf16)x[((size_t)b * kC + c) * kL + l];
}

// ---------------------------------------------------------------------------
// Kernel 2: weights f32 -> bf16, concatenated [wq | wk | wv | wo]
// ---------------------------------------------------------------------------
__global__ void cvt_w_kernel(const float* __restrict__ wq,
                             const float* __restrict__ wk,
                             const float* __restrict__ wv,
                             const float* __restrict__ wo,
                             __bf16* __restrict__ wb) {
  int i = blockIdx.x * blockDim.x + threadIdx.x;   // 0..65535
  wb[i]             = (__bf16)wq[i];
  wb[65536 + i]     = (__bf16)wk[i];
  wb[2 * 65536 + i] = (__bf16)wv[i];
  wb[3 * 65536 + i] = (__bf16)wo[i];
}

// ---------------------------------------------------------------------------
// Kernel 3: fused QKV projection.
//   Y[b][o][l] = sum_c W[o][c] * X[b][c][l] (+bias, q scaled)
// Each wave computes a 16(o) x 16(l) tile with 8 WMMA K=32 steps.
// Outputs: Qt/Kt [b][h][l][32] bf16, Vd [b][h][32][L] bf16.
// grid (16, 64, 12), block 128 (4 waves; wave -> one of 48 (proj,o-tile)).
// ---------------------------------------------------------------------------
__global__ void __launch_bounds__(128)
qkv_proj_kernel(const __bf16* __restrict__ xb_t, const __bf16* __restrict__ wb,
                const float* __restrict__ bq, const float* __restrict__ bk,
                const float* __restrict__ bv, __bf16* __restrict__ Qt,
                __bf16* __restrict__ Kt, __bf16* __restrict__ Vd) {
  const int lane = threadIdx.x & 31;
  const int wave = threadIdx.x >> 5;
  const int h = lane >> 4;      // lane half (0/1)
  const int n = lane & 15;      // A row / B col within tile
  const int b  = blockIdx.x;    // batch
  const int lt = blockIdx.y;    // l tile 0..63
  const int zt = blockIdx.z * 4 + wave; // 0..47
  const int proj = zt >> 4;     // 0=q,1=k,2=v
  const int ot   = zt & 15;     // o tile 0..15

  const __bf16* W = wb + (size_t)proj * 65536;
  const int orow = ot * 16 + n;       // A-fragment row (output channel)
  const int l0 = lt * 16;
  const __bf16* xrow = xb_t + ((size_t)b * kL + l0 + n) * kC; // B col position

  v8f acc = {};
#pragma unroll
  for (int c0 = 0; c0 < kC; c0 += 32) {
    // A: W[orow][c0 + contiguous K]  (elems0-15: K=16h+e)
    const __bf16* ap = W + (size_t)orow * kC + c0 + h * 16;
    v16bf a = make_afrag(ap, ap + 8);
    // B: X[c0+16h+e][l0+n] == xb_t[l0+n][c0+16h+e]  (contiguous)
    v16bf bf = *(const v16bf*)(xrow + c0 + h * 16);
    acc = WMMA_BF16(a, bf, acc);
  }

  const float* bias = (proj == 0) ? bq : ((proj == 1) ? bk : bv);
  const float scale = (proj == 0) ? 0.17677669529663687f : 1.0f; // 32^-0.5

#pragma unroll
  for (int v = 0; v < 8; ++v) {
    const int o = ot * 16 + h * 8 + v;  // D row
    const int l = l0 + n;               // D col
    const float val = (acc[v] + bias[o]) * scale;
    const int hd = o >> 5, d = o & 31;
    const __bf16 bvv = (__bf16)val;
    if (proj == 0)
      Qt[(((size_t)b * kNH + hd) * kL + l) * kDH + d] = bvv;
    else if (proj == 1)
      Kt[(((size_t)b * kNH + hd) * kL + l) * kDH + d] = bvv;
    else
      Vd[(((size_t)b * kNH + hd) * kDH + d) * kL + l] = bvv;
  }
}

// ---------------------------------------------------------------------------
// Kernel 4: flash-style attention per (b, head). One wave per 16-row tile,
// streaming 32 key/value columns per iteration:
//   2x WMMA for S = Q^T K, online softmax via LDS, 2x WMMA for P V.
// Output Ot [b][l][256] bf16 (c = head*32 + d).
// grid (16, 8, 16), block 128 (4 independent waves, 4 row tiles / block).
// ---------------------------------------------------------------------------
__global__ void __launch_bounds__(128)
attn_kernel(const __bf16* __restrict__ Qt, const __bf16* __restrict__ Kt,
            const __bf16* __restrict__ Vd, __bf16* __restrict__ Ot) {
  __shared__ __align__(32) float  Pf[4][16][32];
  __shared__ __align__(32) __bf16 Pb[4][16][32];
  __shared__ float alpha_s[4][16];
  __shared__ float lsum_s[4][16];

  const int lane = threadIdx.x & 31;
  const int wave = threadIdx.x >> 5;
  const int h = lane >> 4;
  const int n = lane & 15;
  const int b  = blockIdx.x;
  const int hd = blockIdx.y;
  const int rt = blockIdx.z * 4 + wave;   // row tile 0..63
  const int l0 = rt * 16;

  const __bf16* Qbase = Qt + ((size_t)b * kNH + hd) * kL * kDH;
  const __bf16* Kbase = Kt + ((size_t)b * kNH + hd) * kL * kDH;
  const __bf16* Vbase = Vd + ((size_t)b * kNH + hd) * kDH * kL;

  // Persistent A-fragment of Q: A[m=l][k=d], row = l0+n.
  const __bf16* qp = Qbase + (size_t)(l0 + n) * kDH + h * 16;
  const v16bf aq = make_afrag(qp, qp + 8);

  v8f o0 = {}, o1 = {};                // PV accumulators (d 0..15 / 16..31)
  float m_r = -1e30f, l_r = 0.0f;      // per-row stats (valid in lanes 0..15)

  for (int j = 0; j < kL; j += 32) {
    // ---- S = Q^T K for columns [j, j+32) -------------------------------
    v16bf bk0 = *(const v16bf*)(Kbase + (size_t)(j + n) * kDH + h * 16);
    v16bf bk1 = *(const v16bf*)(Kbase + (size_t)(j + 16 + n) * kDH + h * 16);
    v8f s0 = {}, s1 = {};
    s0 = WMMA_BF16(aq, bk0, s0);
    s1 = WMMA_BF16(aq, bk1, s1);

    // Spill scores to LDS (D layout: row = 8h+v, col = n / 16+n).
#pragma unroll
    for (int v = 0; v < 8; ++v) {
      Pf[wave][h * 8 + v][n]      = s0[v];
      Pf[wave][h * 8 + v][16 + n] = s1[v];
    }
    asm volatile("s_wait_dscnt 0" ::: "memory");

    // ---- online softmax, lanes 0..15 own one row each ------------------
    if (lane < 16) {
      float cm = m_r;
#pragma unroll
      for (int t = 0; t < 32; ++t) cm = fmaxf(cm, Pf[wave][lane][t]);
      float sum = 0.0f;
#pragma unroll
      for (int t = 0; t < 32; ++t) {
        float p = __expf(Pf[wave][lane][t] - cm);
        sum += p;
        Pb[wave][lane][t] = (__bf16)p;
      }
      const float al = __expf(m_r - cm);
      alpha_s[wave][lane] = al;
      l_r = l_r * al + sum;
      m_r = cm;
    }
    asm volatile("s_wait_dscnt 0" ::: "memory");

    // Rescale running accumulators by per-row alpha.
#pragma unroll
    for (int v = 0; v < 8; ++v) {
      const float al = alpha_s[wave][h * 8 + v];
      o0[v] *= al;
      o1[v] *= al;
    }

    // A-fragment of P from LDS (row = n, contiguous K over 32 cols).
    const __bf16* pr = &Pb[wave][n][0] + h * 16;
    v16bf ap = make_afrag(pr, pr + 8);

    // B-fragments of V: B[k=m][n=d] = Vd[d][j + 16h + e] (contiguous).
    v16bf bv0 = *(const v16bf*)(Vbase + (size_t)n * kL + j + h * 16);
    v16bf bv1 = *(const v16bf*)(Vbase + (size_t)(16 + n) * kL + j + h * 16);

    o0 = WMMA_BF16(ap, bv0, o0);
    o1 = WMMA_BF16(ap, bv1, o1);
  }

  if (lane < 16) lsum_s[wave][lane] = l_r;
  asm volatile("s_wait_dscnt 0" ::: "memory");

  // Normalize and store: Ot[b][l][head*32 + d], D rows = l_local, cols = d.
#pragma unroll
  for (int v = 0; v < 8; ++v) {
    const float inv = 1.0f / lsum_s[wave][h * 8 + v];
    const int l = l0 + h * 8 + v;
    __bf16* orow = Ot + ((size_t)b * kL + l) * kC + hd * kDH;
    orow[n]      = (__bf16)(o0[v] * inv);
    orow[16 + n] = (__bf16)(o1[v] * inv);
  }
}

// ---------------------------------------------------------------------------
// Kernel 5: output projection. Y[b][o][l] = sum_c Wo[o][c] Ot[b][l][c] + bo.
// grid (16, 64, 4), block 128.
// ---------------------------------------------------------------------------
__global__ void __launch_bounds__(128)
out_proj_kernel(const __bf16* __restrict__ Ot, const __bf16* __restrict__ Wo,
                const float* __restrict__ bo, float* __restrict__ y) {
  const int lane = threadIdx.x & 31;
  const int wave = threadIdx.x >> 5;
  const int h = lane >> 4;
  const int n = lane & 15;
  const int b  = blockIdx.x;
  const int lt = blockIdx.y;
  const int ot = blockIdx.z * 4 + wave;   // 0..15

  const int orow = ot * 16 + n;
  const int l0 = lt * 16;
  const __bf16* xrow = Ot + ((size_t)b * kL + l0 + n) * kC;

  v8f acc = {};
#pragma unroll
  for (int c0 = 0; c0 < kC; c0 += 32) {
    const __bf16* ap = Wo + (size_t)orow * kC + c0 + h * 16;
    v16bf a = make_afrag(ap, ap + 8);
    v16bf bf = *(const v16bf*)(xrow + c0 + h * 16);
    acc = WMMA_BF16(a, bf, acc);
  }

#pragma unroll
  for (int v = 0; v < 8; ++v) {
    const int o = ot * 16 + h * 8 + v;
    const int l = l0 + n;
    y[((size_t)b * kC + o) * kL + l] = acc[v] + bo[o];
  }
}

// ---------------------------------------------------------------------------
// Host launcher
// ---------------------------------------------------------------------------
extern "C" void kernel_launch(void* const* d_in, const int* in_sizes, int n_in,
                              void* d_out, int out_size, void* d_ws,
                              size_t ws_size, hipStream_t stream) {
  const float* x  = (const float*)d_in[0];
  const float* wq = (const float*)d_in[1];
  const float* bq = (const float*)d_in[2];
  const float* wk = (const float*)d_in[3];
  const float* bk = (const float*)d_in[4];
  const float* wv = (const float*)d_in[5];
  const float* bv = (const float*)d_in[6];
  const float* wo = (const float*)d_in[7];
  const float* bo = (const float*)d_in[8];
  float* y = (float*)d_out;

  char* ws = (char*)d_ws;
  size_t off = 0;
  __bf16* xb_t = (__bf16*)(ws + off); off += (size_t)kB * kL * kC * 2;      // 8 MB
  __bf16* wb   = (__bf16*)(ws + off); off += (size_t)4 * kC * kC * 2;       // 512 KB
  __bf16* Qt   = (__bf16*)(ws + off); off += (size_t)kB * kNH * kL * kDH * 2;
  __bf16* Kt   = (__bf16*)(ws + off); off += (size_t)kB * kNH * kL * kDH * 2;
  __bf16* Vd   = (__bf16*)(ws + off); off += (size_t)kB * kNH * kDH * kL * 2;
  __bf16* Ot   = (__bf16*)(ws + off); off += (size_t)kB * kL * kC * 2;

  cvt_x_kernel<<<dim3((kB * kL * kC) / 256), 256, 0, stream>>>(x, xb_t);
  cvt_w_kernel<<<dim3(65536 / 256), 256, 0, stream>>>(wq, wk, wv, wo, wb);
  qkv_proj_kernel<<<dim3(16, 64, 12), 128, 0, stream>>>(xb_t, wb, bq, bk, bv,
                                                        Qt, Kt, Vd);
  attn_kernel<<<dim3(16, 8, 16), 128, 0, stream>>>(Qt, Kt, Vd, Ot);
  out_proj_kernel<<<dim3(16, 64, 4), 128, 0, stream>>>(Ot, wb + 3 * 65536, bo,
                                                       y);
}
